// GCNNet_45749991637433
// MI455X (gfx1250) — matrix operand order repaired and checked by reading, not compiled, not run on hardware
//
#include <hip/hip_runtime.h>
#include <hip/hip_bf16.h>

typedef __attribute__((ext_vector_type(16))) __bf16    v16bf;
typedef __attribute__((ext_vector_type(8)))  float     v8f;
typedef __attribute__((ext_vector_type(8)))  unsigned  v8u;
typedef __attribute__((ext_vector_type(4)))  unsigned  v4u;

// ---------- helpers ----------
__device__ __forceinline__ unsigned short f2bf_bits(float f) {
    unsigned u = __builtin_bit_cast(unsigned, f);
    return (unsigned short)((u + 0x7FFFu + ((u >> 16) & 1u)) >> 16);  // RNE
}
__device__ __forceinline__ void atomAddF(float* p, float v) {
    unsafeAtomicAdd(p, v);   // global_atomic_add_f32
}

// ---------- degree / normalization ----------
__global__ void k_set_ones(float* deg, int n) {
    int i = blockIdx.x * blockDim.x + threadIdx.x;
    if (i < n) deg[i] = 1.0f;                 // self-loop contributes 1
}
__global__ void k_deg_count(const int* __restrict__ dst, float* deg, int E) {
    int e = blockIdx.x * blockDim.x + threadIdx.x;
    if (e < E) atomAddF(&deg[dst[e]], 1.0f);
}
__global__ void k_rsqrt(float* d, int n) {
    int i = blockIdx.x * blockDim.x + threadIdx.x;
    if (i < n) d[i] = rsqrtf(d[i]);
}

// ---------- pack f32 -> bf16 (optionally fused ReLU), 2 elements/thread ----------
__global__ void k_pack_bf16(const float* __restrict__ h, unsigned* __restrict__ p,
                            int n2, int relu) {
    int i = blockIdx.x * blockDim.x + threadIdx.x;
    if (i >= n2) return;
    float f0 = h[2 * i], f1 = h[2 * i + 1];
    if (relu) { f0 = fmaxf(f0, 0.0f); f1 = fmaxf(f1, 0.0f); }
    p[i] = (unsigned)f2bf_bits(f0) | ((unsigned)f2bf_bits(f1) << 16);
}

// ---------- weight transpose + convert: Wt[c*K+k] = bf16(W[k*C+c]) ----------
__global__ void k_wt_bf16(const float* __restrict__ W, unsigned short* __restrict__ Wt,
                          int K, int C) {
    int i = blockIdx.x * blockDim.x + threadIdx.x;
    if (i >= K * C) return;
    int c = i / K, k = i - c * K;
    Wt[i] = f2bf_bits(W[(size_t)k * C + c]);
}

// ---------- WMMA GEMM: Out[N x C] = A_bf16[N x K] @ Wt_bf16[C x K]^T ----------
// One wave per 16x64 output tile (4 WMMA column tiles). K%32==0, C%64==0, N%16==0.
__global__ void k_gemm_wmma(const __bf16* __restrict__ A, const __bf16* __restrict__ Wt,
                            float* __restrict__ Out, int K, int C) {
    const int lane = threadIdx.x;        // 0..31
    const int m16  = lane & 15;
    const int half = lane >> 4;
    const int tileM   = blockIdx.x;
    const int colBase = blockIdx.y * 64;

    const __bf16* arow = A + (size_t)(tileM * 16 + m16) * K;
    const __bf16* wrow0 = Wt + (size_t)(colBase + m16) * K;        // tile t adds t*16*K

    v8f acc0 = {}, acc1 = {}, acc2 = {}, acc3 = {};
    for (int k0 = 0; k0 < K; k0 += 32) {
        // A fragment: 32 contiguous bytes per lane, k = 16*half + s (matches B)
        v4u alo = *(const v4u*)(arow + k0 + half * 16);
        v4u ahi = *(const v4u*)(arow + k0 + half * 16 + 8);
        v8u au;
        #pragma unroll
        for (int j = 0; j < 4; ++j) { au[j] = alo[j]; au[4 + j] = ahi[j]; }
        v16bf a = __builtin_bit_cast(v16bf, au);

        // B fragments: 32 contiguous bytes of Wt[col] per lane
        #pragma unroll
        for (int t = 0; t < 4; ++t) {
            v8u bu = *(const v8u*)(wrow0 + (size_t)t * 16 * K + k0 + half * 16);
            v16bf b = __builtin_bit_cast(v16bf, bu);
            v8f& acc = (t == 0) ? acc0 : (t == 1) ? acc1 : (t == 2) ? acc2 : acc3;
            acc = __builtin_amdgcn_wmma_f32_16x16x32_bf16(
                      false, a, false, b, (short)0, acc, false, false);
        }
    }
    // D layout: VGPR i -> row i + 8*half, col m16
    float* orow = Out + (size_t)(tileM * 16 + half * 8) * C + colBase + m16;
    #pragma unroll
    for (int t = 0; t < 4; ++t) {
        const v8f& acc = (t == 0) ? acc0 : (t == 1) ? acc1 : (t == 2) ? acc2 : acc3;
        #pragma unroll
        for (int i = 0; i < 8; ++i)
            orow[(size_t)i * C + t * 16] = acc[i];
    }
}

// ---------- aggregation: out[v] = t[v]*dinv[v]^2 + b   (self-loop + bias) ----------
__global__ void k_agg_init(const float* __restrict__ t, const float* __restrict__ dinv,
                           const float* __restrict__ bias, float* __restrict__ out,
                           int n, int C) {
    int i = blockIdx.x * blockDim.x + threadIdx.x;
    if (i >= n * C) return;
    int v = i / C, c = i - v * C;
    float w = dinv[v];
    out[i] = t[i] * w * w + bias[c];
}

// ---------- aggregation: edge scatter-add, one wave per edge ----------
__global__ void k_agg_edges(const float* __restrict__ t, const int* __restrict__ src,
                            const int* __restrict__ dst, const float* __restrict__ dinv,
                            float* __restrict__ out, int E, int C) {
    int e = blockIdx.x * (blockDim.x >> 5) + (threadIdx.x >> 5);
    int lane = threadIdx.x & 31;
    if (e >= E) return;
    int s = src[e], d = dst[e];
    float w = dinv[s] * dinv[d];
    const float* ts = t + (size_t)s * C;
    float* od = out + (size_t)d * C;
    for (int c = lane; c < C; c += 32)
        atomAddF(&od[c], ts[c] * w);
}

// ---------- layer 5: t[v, 0:2] = relu(h[v]) @ W5 (K=64, C=2) ----------
__global__ void k_dense2(const float* __restrict__ h, const float* __restrict__ W,
                         float* __restrict__ t, int n) {
    int v = blockIdx.x * blockDim.x + threadIdx.x;
    if (v >= n) return;
    const float* hr = h + (size_t)v * 64;
    float a0 = 0.0f, a1 = 0.0f;
    #pragma unroll 8
    for (int k = 0; k < 64; ++k) {
        float x = fmaxf(hr[k], 0.0f);
        a0 += x * W[2 * k];
        a1 += x * W[2 * k + 1];
    }
    t[2 * v] = a0;
    t[2 * v + 1] = a1;
}

__global__ void k_agg_edges2(const float* __restrict__ t, const int* __restrict__ src,
                             const int* __restrict__ dst, const float* __restrict__ dinv,
                             float* __restrict__ out, int E) {
    int i = blockIdx.x * blockDim.x + threadIdx.x;
    if (i >= 2 * E) return;
    int e = i >> 1, c = i & 1;
    int s = src[e], d = dst[e];
    atomAddF(&out[2 * d + c], t[2 * s + c] * dinv[s] * dinv[d]);
}

__global__ void k_logsoftmax2(float* __restrict__ out, int n) {
    int v = blockIdx.x * blockDim.x + threadIdx.x;
    if (v >= n) return;
    float z0 = out[2 * v], z1 = out[2 * v + 1];
    float m = fmaxf(z0, z1);
    float l = m + logf(expf(z0 - m) + expf(z1 - m));
    out[2 * v] = z0 - l;
    out[2 * v + 1] = z1 - l;
}

// ---------- host launch ----------
extern "C" void kernel_launch(void* const* d_in, const int* in_sizes, int n_in,
                              void* d_out, int out_size, void* d_ws, size_t ws_size,
                              hipStream_t stream) {
    const float* x   = (const float*)d_in[0];
    const int*   ei  = (const int*)d_in[1];
    const float* W1  = (const float*)d_in[2];
    const float* b1  = (const float*)d_in[3];
    const float* W2  = (const float*)d_in[4];
    const float* b2  = (const float*)d_in[5];
    const float* W3  = (const float*)d_in[6];
    const float* b3  = (const float*)d_in[7];
    const float* W4  = (const float*)d_in[8];
    const float* b4  = (const float*)d_in[9];
    const float* W5  = (const float*)d_in[10];
    const float* b5  = (const float*)d_in[11];

    const int N = in_sizes[0] / 128;       // 50000
    const int E = in_sizes[1] / 2;         // 400000
    const int* src = ei;
    const int* dst = ei + E;

    // ---- workspace carve-up (all 256B-aligned) ----
    char* ws = (char*)d_ws;
    size_t off = 0;
    auto carve = [&](size_t bytes) { char* p = ws + off;
                                     off += (bytes + 255) & ~(size_t)255; return p; };
    float*          dinv = (float*)carve((size_t)N * 4);
    float*          bufT = (float*)carve((size_t)N * 512 * 4);   // GEMM out
    float*          bufA = (float*)carve((size_t)N * 256 * 4);   // agg out (odd layers)
    float*          bufB = (float*)carve((size_t)N * 512 * 4);   // agg out (even layers)
    __bf16*         bufP = (__bf16*)carve((size_t)N * 512 * 2);  // packed GEMM input
    unsigned short* wt1  = (unsigned short*)carve((size_t)128 * 256 * 2);
    unsigned short* wt2  = (unsigned short*)carve((size_t)256 * 512 * 2);
    unsigned short* wt3  = (unsigned short*)carve((size_t)512 * 256 * 2);
    unsigned short* wt4  = (unsigned short*)carve((size_t)256 * 64 * 2);
    (void)ws_size; (void)n_in; (void)out_size;

    float* out2 = (float*)d_out;                                 // N x 2

    const int TB = 256;
    // ---- normalization ----
    k_set_ones <<<(N + TB - 1) / TB, TB, 0, stream>>>(dinv, N);
    k_deg_count<<<(E + TB - 1) / TB, TB, 0, stream>>>(dst, dinv, E);
    k_rsqrt    <<<(N + TB - 1) / TB, TB, 0, stream>>>(dinv, N);

    // ---- weight transpose/convert (tiny, once per call) ----
    k_wt_bf16<<<(128 * 256 + TB - 1) / TB, TB, 0, stream>>>(W1, wt1, 128, 256);
    k_wt_bf16<<<(256 * 512 + TB - 1) / TB, TB, 0, stream>>>(W2, wt2, 256, 512);
    k_wt_bf16<<<(512 * 256 + TB - 1) / TB, TB, 0, stream>>>(W3, wt3, 512, 256);
    k_wt_bf16<<<(256 * 64  + TB - 1) / TB, TB, 0, stream>>>(W4, wt4, 256, 64);

    const int edgeBlocks = (E + 7) / 8;    // 8 waves/block, one edge each

    // ---- layer 1: x(128) -> 256 ----
    k_pack_bf16<<<(N * 128 / 2 + TB - 1) / TB, TB, 0, stream>>>(x, (unsigned*)bufP, N * 64, 0);
    k_gemm_wmma<<<dim3(N / 16, 256 / 64), 32, 0, stream>>>(bufP, (const __bf16*)wt1, bufT, 128, 256);
    k_agg_init <<<(N * 256 + TB - 1) / TB, TB, 0, stream>>>(bufT, dinv, b1, bufA, N, 256);
    k_agg_edges<<<edgeBlocks, TB, 0, stream>>>(bufT, src, dst, dinv, bufA, E, 256);

    // ---- layer 2: 256 -> 512 (relu folded into pack) ----
    k_pack_bf16<<<(N * 256 / 2 + TB - 1) / TB, TB, 0, stream>>>(bufA, (unsigned*)bufP, N * 128, 1);
    k_gemm_wmma<<<dim3(N / 16, 512 / 64), 32, 0, stream>>>(bufP, (const __bf16*)wt2, bufT, 256, 512);
    k_agg_init <<<(N * 512 + TB - 1) / TB, TB, 0, stream>>>(bufT, dinv, b2, bufB, N, 512);
    k_agg_edges<<<edgeBlocks, TB, 0, stream>>>(bufT, src, dst, dinv, bufB, E, 512);

    // ---- layer 3: 512 -> 256 ----
    k_pack_bf16<<<(N * 512 / 2 + TB - 1) / TB, TB, 0, stream>>>(bufB, (unsigned*)bufP, N * 256, 1);
    k_gemm_wmma<<<dim3(N / 16, 256 / 64), 32, 0, stream>>>(bufP, (const __bf16*)wt3, bufT, 512, 256);
    k_agg_init <<<(N * 256 + TB - 1) / TB, TB, 0, stream>>>(bufT, dinv, b3, bufA, N, 256);
    k_agg_edges<<<edgeBlocks, TB, 0, stream>>>(bufT, src, dst, dinv, bufA, E, 256);

    // ---- layer 4: 256 -> 64 ----
    k_pack_bf16<<<(N * 256 / 2 + TB - 1) / TB, TB, 0, stream>>>(bufA, (unsigned*)bufP, N * 128, 1);
    k_gemm_wmma<<<dim3(N / 16, 64 / 64), 32, 0, stream>>>(bufP, (const __bf16*)wt4, bufT, 256, 64);
    k_agg_init <<<(N * 64 + TB - 1) / TB, TB, 0, stream>>>(bufT, dinv, b4, bufB, N, 64);
    k_agg_edges<<<edgeBlocks, TB, 0, stream>>>(bufT, src, dst, dinv, bufB, E, 64);

    // ---- layer 5: 64 -> 2, then log_softmax ----
    k_dense2    <<<(N + TB - 1) / TB, TB, 0, stream>>>(bufB, W5, bufT, N);
    k_agg_init  <<<(N * 2 + TB - 1) / TB, TB, 0, stream>>>(bufT, dinv, b5, out2, N, 2);
    k_agg_edges2<<<(2 * E + TB - 1) / TB, TB, 0, stream>>>(bufT, src, dst, dinv, out2, E);
    k_logsoftmax2<<<(N + TB - 1) / TB, TB, 0, stream>>>(out2, N);
}
